// HistCausalEncoder_71691594105230
// MI455X (gfx1250) — matrix operand order repaired and checked by reading, not compiled, not run on hardware
//
#include <hip/hip_runtime.h>

// ---------------- problem constants (from reference) ----------------
#define BB   16          // batch
#define TT   64          // time steps
#define PP   64          // particles
#define INC  256         // o feature dim
#define ACTC 64          // act feature dim
#define OUTC 256         // output dim
#define HH   4           // history slots
#define SLOT (OUTC + ACTC)        // 320
#define CONCAT (INC + HH * SLOT)  // 1536
#define KB32 (CONCAT / 32)        // 48 k-blocks of 32
#define MROWS 16                  // (b,p) rows per workgroup
#define XROW 1544                 // padded LDS row stride in bf16 elems (772 dwords -> 4-bank rotate/row)

typedef __bf16 v16bf __attribute__((ext_vector_type(16)));
typedef __bf16 v8bf  __attribute__((ext_vector_type(8)));
typedef float  v8f   __attribute__((ext_vector_type(8)));

static __device__ __forceinline__ unsigned short bf16bits(float f) {
    unsigned u = __builtin_bit_cast(unsigned, f);
    u += 0x7fffu + ((u >> 16) & 1u);           // round-to-nearest-even
    return (unsigned short)(u >> 16);
}

// A fragment (16x32 bf16): 32 contiguous bytes per lane
static __device__ __forceinline__ v16bf loadA(const unsigned short* p) {
    v8bf lo = *(const v8bf*)(p);
    v8bf hi = *(const v8bf*)(p + 8);
    v16bf a;
    #pragma unroll
    for (int j = 0; j < 8; ++j) { a[j] = lo[j]; a[8 + j] = hi[j]; }
    return a;
}

// B fragment: 32 contiguous bytes per lane (fragment-linear swizzled W)
static __device__ __forceinline__ v16bf loadB(const unsigned short* p) {
    v8bf lo = *(const v8bf*)(p);
    v8bf hi = *(const v8bf*)(p + 8);
    v16bf b;
    #pragma unroll
    for (int j = 0; j < 8; ++j) { b[j] = lo[j]; b[8 + j] = hi[j]; }
    return b;
}

// ---------------------------------------------------------------
// Prep: W (OUTC x CONCAT fp32, row major) -> bf16, swizzled into
// B-fragment-linear layout for v_wmma_f32_16x16x32_bf16.
// Fragment (nt, kb): lane L holds 16 contiguous bf16 =
//   W[nt*16 + L%16][kb*32 + (L/16)*16 + 0..15]
// ---------------------------------------------------------------
__global__ void __launch_bounds__(256)
wprep_kernel(const float* __restrict__ W, unsigned short* __restrict__ ws) {
    int f    = blockIdx.x * 256 + threadIdx.x;   // 0 .. 393215
    int j    = f & 15;
    int lane = (f >> 4) & 31;
    int frag = f >> 9;
    int kb   = frag % KB32;
    int nt   = frag / KB32;
    int n    = nt * 16 + (lane & 15);
    int k    = kb * 32 + (lane >> 4) * 16 + j;
    ws[f] = bf16bits(W[n * CONCAT + k]);
}

// ---------------------------------------------------------------
// Main recurrent kernel: one WG = 16 (b,p) rows, all T steps.
// __launch_bounds__(256, 2): cap VGPRs at 512 so 8 waves = 2/SIMD fit.
// ---------------------------------------------------------------
__global__ void __launch_bounds__(256, 2)
hist_main_kernel(const float* __restrict__ o,
                 const float* __restrict__ act,
                 const float* __restrict__ bias,
                 const unsigned short* __restrict__ wsw,
                 float* __restrict__ out) {
    __shared__ unsigned short xbuf[MROWS * XROW];   // 16 rows x 1544 bf16

    const int tid  = threadIdx.x;
    const int wave = tid >> 5;
    const int ln   = tid & 31;
    const int half = ln >> 4;       // 0/1
    const int lm   = ln & 15;       // A row / C column-within-tile
    const int nt0  = wave * 2;
    const int nt1  = wave * 2 + 1;

    const int b0 = blockIdx.x >> 2;         // batch index (16 rows share b)
    const int p0 = (blockIdx.x & 3) * 16;   // first particle of this block

    const float bias0 = bias[nt0 * 16 + lm];
    const float bias1 = bias[nt1 * 16 + lm];

    //  A row base within a slot (add slot base per h)
    const int aRow = lm * XROW + half * 16;

    // zero the history region (cols [0, HH*SLOT) of every row), as dwords
    {
        unsigned int* x32 = (unsigned int*)xbuf;
        const int perRow = (HH * SLOT) / 2;            // 640 dwords
        for (int idx = tid; idx < MROWS * perRow; idx += 256) {
            int m = idx / perRow, c = idx % perRow;
            x32[m * (XROW / 2) + c] = 0u;
        }
    }

    for (int t = 0; t < TT; ++t) {
        const long obase   = ((long)(b0 * TT + t) * PP + p0) * INC;
        const long abase   = ((long)(b0 * TT + t) * PP + p0) * ACTC;
        const long outbase = ((long)(b0 * TT + t) * PP + p0) * OUTC;

        // ---- stage o_t into LDS cols [HH*SLOT, CONCAT), bf16, coalesced ----
        #pragma unroll 4
        for (int i = 0; i < MROWS; ++i) {
            float v = __builtin_nontemporal_load(&o[obase + i * INC + tid]);
            xbuf[i * XROW + HH * SLOT + tid] = bf16bits(v);
            if (t + 1 < TT)   // pull next step's o block toward the caches
                __builtin_prefetch(&o[obase + (long)PP * INC + i * INC + tid], 0, 1);
        }
        __syncthreads();   // staging + previous step's history writes visible

        // Defeat LICM: without this, the (t-invariant) B-fragment loads get
        // hoisted out of the t-loop, blow past the register file and spill to
        // scratch. W is L2-resident (768 KB shared by all WGs) — re-loading it
        // each step from L2 is the fast path.
        asm volatile("" ::: "memory");

        // ---- GEMM: acc(16x16) per n-tile, K = 4 slots x 10 + o-region x 8 ----
        v8f c0 = {};
        v8f c1 = {};
        #pragma unroll
        for (int h = 0; h < HH; ++h) {
            const unsigned short* ap  = &xbuf[aRow + ((t + h) & 3) * SLOT];
            const unsigned short* bp0 = wsw + ((unsigned)(nt0 * KB32 * 32 + ln)) * 16 + (h * 10) * 512;
            const unsigned short* bp1 = wsw + ((unsigned)(nt1 * KB32 * 32 + ln)) * 16 + (h * 10) * 512;
            #pragma unroll 2
            for (int w = 0; w < SLOT / 32; ++w) {      // 10 K-blocks
                v16bf a  = loadA(ap + w * 32);
                v16bf b0 = loadB(bp0 + w * 512);
                v16bf b1 = loadB(bp1 + w * 512);
                c0 = __builtin_amdgcn_wmma_f32_16x16x32_bf16(
                         false, a, false, b0, (short)0, c0, false, false);
                c1 = __builtin_amdgcn_wmma_f32_16x16x32_bf16(
                         false, a, false, b1, (short)0, c1, false, false);
            }
        }
        {
            const unsigned short* ap  = &xbuf[aRow + HH * SLOT];
            const unsigned short* bp0 = wsw + ((unsigned)(nt0 * KB32 * 32 + ln)) * 16 + 40 * 512;
            const unsigned short* bp1 = wsw + ((unsigned)(nt1 * KB32 * 32 + ln)) * 16 + 40 * 512;
            #pragma unroll 2
            for (int w = 0; w < INC / 32; ++w) {       // 8 K-blocks (o_t region)
                v16bf a  = loadA(ap + w * 32);
                v16bf b0 = loadB(bp0 + w * 512);
                v16bf b1 = loadB(bp1 + w * 512);
                c0 = __builtin_amdgcn_wmma_f32_16x16x32_bf16(
                         false, a, false, b0, (short)0, c0, false, false);
                c1 = __builtin_amdgcn_wmma_f32_16x16x32_bf16(
                         false, a, false, b1, (short)0, c1, false, false);
            }
        }
        __syncthreads();   // all waves done reading slot (t&3) before rewrite

        // ---- write z_t: global fp32 (streaming NT) + LDS history slot (bf16) ----
        {
            const int sc = (t & 3) * SLOT;
            #pragma unroll
            for (int j = 0; j < 8; ++j) {
                int m = half * 8 + j;            // C layout: VGPR j -> M = half*8+j, N = lm
                float z0 = c0[j] + bias0;
                float z1 = c1[j] + bias1;
                __builtin_nontemporal_store(z0, &out[outbase + m * OUTC + nt0 * 16 + lm]);
                __builtin_nontemporal_store(z1, &out[outbase + m * OUTC + nt1 * 16 + lm]);
                xbuf[m * XROW + sc + nt0 * 16 + lm] = bf16bits(z0);
                xbuf[m * XROW + sc + nt1 * 16 + lm] = bf16bits(z1);
            }
            // ---- stage a_t into slot cols [256, 320) ----
            #pragma unroll
            for (int i = 0; i < 4; ++i) {
                int idx = tid + i * 256;
                int m = idx >> 6, c = idx & 63;
                float av = __builtin_nontemporal_load(&act[abase + m * ACTC + c]);
                xbuf[m * XROW + sc + OUTC + c] = bf16bits(av);
            }
        }
        // next iteration's barrier orders these writes before the next GEMM
    }
}

extern "C" void kernel_launch(void* const* d_in, const int* in_sizes, int n_in,
                              void* d_out, int out_size, void* d_ws, size_t ws_size,
                              hipStream_t stream) {
    const float* o    = (const float*)d_in[0];
    const float* act  = (const float*)d_in[1];
    const float* W    = (const float*)d_in[2];
    const float* bias = (const float*)d_in[3];
    float* out = (float*)d_out;
    unsigned short* ws = (unsigned short*)d_ws;   // 256*1536 bf16 = 768 KB

    // 1) convert + swizzle W into B-fragment-linear bf16
    wprep_kernel<<<(OUTC * CONCAT) / 256, 256, 0, stream>>>(W, ws);

    // 2) recurrent WMMA kernel: 64 WGs x 16 rows, all 64 steps
    hist_main_kernel<<<(BB * PP) / MROWS, 256, 0, stream>>>(o, act, bias, ws, out);
}
